// ContextGatingSigmoidClassifier_32959579030184
// MI455X (gfx1250) — matrix operand order repaired and checked by reading, not compiled, not run on hardware
//
#include <hip/hip_runtime.h>

typedef _Float16 v8h  __attribute__((ext_vector_type(8)));
typedef _Float16 v16h __attribute__((ext_vector_type(16)));
typedef float    v8f  __attribute__((ext_vector_type(8)));

#define BN_EPS    1e-5f
#define NEG_SLOPE 0.2f

static constexpr int BQ   = 32;    // batch
static constexpr int CIN  = 2048;  // input channels per branch
static constexpr int NQ   = 588;   // N*H*W = 12*7*7 pixels per batch
static constexpr int NCH  = 512;   // branch output channels
static constexpr int NCAT = 1024;  // concat channels
static constexpr int NH   = 256;   // hidden
static constexpr int NSEG = 12;    // N
static constexpr int HW   = 49;    // H*W
static constexpr int LDF  = 592;   // padded row stride (halfs) for f16 intermediate (16B aligned)

// Largest 8-aligned column base so a 16-wide staging group stays inside [0, NQ)
static constexpr int QCLAMP = (NQ - 16) & ~7;   // 568

// ---------------------------------------------------------------------------
// Wait helper for async global->LDS copies
// ---------------------------------------------------------------------------
__device__ __forceinline__ void wait_async0() {
#if __has_builtin(__builtin_amdgcn_s_wait_asynccnt)
    __builtin_amdgcn_s_wait_asynccnt(0);
#else
    asm volatile("s_wait_asynccnt 0" ::: "memory");
#endif
}

// ---------------------------------------------------------------------------
// Block reduction helper (256 threads)
// ---------------------------------------------------------------------------
__device__ __forceinline__ float block_reduce_sum(float v) {
    __shared__ float sm[256];
    sm[threadIdx.x] = v;
    __syncthreads();
    for (int s = 128; s > 0; s >>= 1) {
        if ((int)threadIdx.x < s) sm[threadIdx.x] += sm[threadIdx.x + s];
        __syncthreads();
    }
    return sm[0];
}

// ---------------------------------------------------------------------------
// BN-fold prep kernels: produce f16 weights (BN scale folded in) + f32 bias
// ---------------------------------------------------------------------------
__global__ __launch_bounds__(256) void fold_branch_kernel(
    const float* __restrict__ W,   // [O, Cin]
    const float* __restrict__ g, const float* __restrict__ bb,
    const float* __restrict__ mm, const float* __restrict__ vv,
    const float* __restrict__ blin,
    _Float16* __restrict__ W16, float* __restrict__ bout, int Cin)
{
    const int o = blockIdx.x;
    float partial = 0.f;
    for (int c = threadIdx.x; c < Cin; c += 256) {
        float s = g[c] * rsqrtf(vv[c] + BN_EPS);
        float w = W[(size_t)o * Cin + c];
        W16[(size_t)o * Cin + c] = (_Float16)(w * s);
        partial += w * (bb[c] - mm[c] * s);
    }
    float tot = block_reduce_sum(partial);
    if (threadIdx.x == 0) bout[o] = blin[o] + tot;
}

__global__ __launch_bounds__(256) void fold1_kernel(
    const float* __restrict__ W1,
    const float* __restrict__ g1, const float* __restrict__ bb1,
    const float* __restrict__ m1, const float* __restrict__ v1,
    const float* __restrict__ b1,
    const float* __restrict__ g2, const float* __restrict__ bb2,
    const float* __restrict__ m2, const float* __restrict__ v2,
    _Float16* __restrict__ W16, float* __restrict__ bout)
{
    const int o = blockIdx.x;
    const float s2 = g2[o] * rsqrtf(v2[o] + BN_EPS);
    float partial = 0.f;
    for (int c = threadIdx.x; c < NCAT; c += 256) {
        float s1 = g1[c] * rsqrtf(v1[c] + BN_EPS);
        float w = W1[(size_t)o * NCAT + c];
        W16[(size_t)o * NCAT + c] = (_Float16)(w * s1 * s2);
        partial += w * (bb1[c] - m1[c] * s1);
    }
    float tot = block_reduce_sum(partial);
    if (threadIdx.x == 0)
        bout[o] = (b1[o] + tot) * s2 + bb2[o] - m2[o] * s2;
}

// ---------------------------------------------------------------------------
// WMMA GEMM core:  D[M x NQ] = A[M x K](f16) * B[K x NQ]  + bias, leaky-ReLU
// Block: 256 threads = 8 waves; tile 128(M) x 64(N); K-step 64.
// Wave grid: 4 (M) x 2 (N); each wave owns 2x2 16x16 f32 accumulators and
// runs 8 v_wmma_f32_16x16x32_f16 per K-step.
// A fragments direct from global (L2-hot weights) in the ISA 16-bit A layout.
// B tile staged in LDS: GEMM1 converts f32->f16 through VGPRs (+prefetch of
// next slab); GEMM2 uses CDNA5 async global->LDS (ASYNCcnt) b128 transfers.
// Column base is clamped so every staging access is a full in-bounds vector
// op; out-of-range output columns are masked at store.
// ---------------------------------------------------------------------------
template <bool BSRC_F32, bool OUT_F16>
__device__ __forceinline__ void gemm_core(
    const _Float16* __restrict__ A,     // [Mtot x K], row-major f16
    const void*     __restrict__ Bsrc,  // [K x ldb], f32 or f16
    const float*    __restrict__ bias,  // [Mtot]
    void*           __restrict__ Out,   // [Mtot x ldo]
    int K, int ldb, int ldo, int mblock, int nblock)
{
    __shared__ _Float16 Bt[64][72];     // 64(K) x 64(N) tile, +8 pad

    const int tid   = threadIdx.x;
    const int lane  = tid & 31;
    const int wave  = tid >> 5;
    const int wm    = wave & 3;         // 0..3  (M groups of 32)
    const int wn    = wave >> 2;        // 0..1  (N groups of 32)
    const int m0    = mblock * 128 + wm * 32;
    const int qbase = nblock * 64;

    const int srow = tid >> 2;          // 0..63 : K row staged by this thread
    const int scol = (tid & 3) << 4;    // 0,16,32,48 : 16-wide column group
    int cq = qbase + scol;              // clamped global column base
    if (cq > QCLAMP) cq = QCLAMP;

    const int mlow = lane & 15;
    const int half = lane >> 4;

    v8f acc[2][2] = {};

    for (int k0 = 0; k0 < K; k0 += 64) {
        // ---------------- stage B tile (64 x 64) into LDS ----------------
        if (BSRC_F32) {
            const float* bp = (const float*)Bsrc + (size_t)(k0 + srow) * ldb + cq;
            const float4 p0 = *(const float4*)(bp + 0);
            const float4 p1 = *(const float4*)(bp + 4);
            const float4 p2 = *(const float4*)(bp + 8);
            const float4 p3 = *(const float4*)(bp + 12);
            v8h t0, t1;
            t0[0] = (_Float16)p0.x; t0[1] = (_Float16)p0.y;
            t0[2] = (_Float16)p0.z; t0[3] = (_Float16)p0.w;
            t0[4] = (_Float16)p1.x; t0[5] = (_Float16)p1.y;
            t0[6] = (_Float16)p1.z; t0[7] = (_Float16)p1.w;
            t1[0] = (_Float16)p2.x; t1[1] = (_Float16)p2.y;
            t1[2] = (_Float16)p2.z; t1[3] = (_Float16)p2.w;
            t1[4] = (_Float16)p3.x; t1[5] = (_Float16)p3.y;
            t1[6] = (_Float16)p3.z; t1[7] = (_Float16)p3.w;
            *(v8h*)&Bt[srow][scol]     = t0;
            *(v8h*)&Bt[srow][scol + 8] = t1;
            if (k0 + 64 < K)
                __builtin_prefetch((const void*)(bp + (size_t)64 * ldb), 0, 0);
        } else {
            const _Float16* bp = (const _Float16*)Bsrc + (size_t)(k0 + srow) * ldb + cq;
            const unsigned  lds0 = (unsigned)(uintptr_t)&Bt[srow][scol];
            const unsigned long long g0 = (unsigned long long)(uintptr_t)bp;
            asm volatile("global_load_async_to_lds_b128 %0, %1, off"
                         :: "v"(lds0), "v"(g0) : "memory");
            asm volatile("global_load_async_to_lds_b128 %0, %1, off offset:16"
                         :: "v"(lds0), "v"(g0) : "memory");
            wait_async0();
        }
        __syncthreads();

        // ---------------- 2 x (2x2) WMMA over the staged slab ------------
#pragma unroll
        for (int kk = 0; kk < 2; ++kk) {
            const int kofs = k0 + kk * 32;

            // A fragments: lane M = lane&15, half = lane>>4,
            // K elements {half*8 + 0..7, 16 + half*8 + 0..7}
            v16h afrag[2];
#pragma unroll
            for (int i = 0; i < 2; ++i) {
                const _Float16* ap =
                    A + (size_t)(m0 + i * 16 + mlow) * K + kofs + half * 8;
                v8h alo = *(const v8h*)ap;
                v8h ahi = *(const v8h*)(ap + 16);
                afrag[i] = __builtin_shufflevector(alo, ahi,
                    0, 1, 2, 3, 4, 5, 6, 7, 8, 9, 10, 11, 12, 13, 14, 15);
            }

            // B fragments: lane N = lane&15, half = lane>>4,
            // K elements {half*8 + 0..7, 16 + half*8 + 0..7}
            v16h bfrag[2];
#pragma unroll
            for (int j = 0; j < 2; ++j) {
                const _Float16* rp = &Bt[kk * 32 + half * 8][wn * 32 + j * 16 + mlow];
#pragma unroll
                for (int s = 0; s < 8; ++s) {
                    bfrag[j][s]     = rp[s * 72];
                    bfrag[j][s + 8] = rp[(16 + s) * 72];
                }
            }

#pragma unroll
            for (int i = 0; i < 2; ++i)
#pragma unroll
                for (int j = 0; j < 2; ++j)
                    acc[i][j] = __builtin_amdgcn_wmma_f32_16x16x32_f16(
                        false, afrag[i], false, bfrag[j],
                        (short)0, acc[i][j], false, false);
        }
        __syncthreads();
    }

    // ---- epilogue: bias + leaky ReLU + store (mask out-of-range cols) ----
#pragma unroll
    for (int j = 0; j < 2; ++j) {
        const int q = qbase + wn * 32 + j * 16 + (lane & 15);
        if (q >= NQ) continue;
#pragma unroll
        for (int i = 0; i < 2; ++i) {
#pragma unroll
            for (int r = 0; r < 8; ++r) {
                const int m = m0 + i * 16 + half * 8 + r;
                float v = acc[i][j][r] + bias[m];
                v = (v >= 0.f) ? v : NEG_SLOPE * v;
                if (OUT_F16)
                    ((_Float16*)Out)[(size_t)m * ldo + q] = (_Float16)v;
                else
                    ((float*)Out)[(size_t)m * ldo + q] = v;
            }
        }
    }
}

// GEMM1: both branches, BN-folded f16 weights, f32 activations -> f16 f-buffer
__global__ __launch_bounds__(256) void gemm1_kernel(
    const _Float16* __restrict__ Wso16, const _Float16* __restrict__ Wc16,
    const float* __restrict__ xso, const float* __restrict__ xc,
    const float* __restrict__ bso, const float* __restrict__ bc,
    _Float16* __restrict__ fbuf)
{
    const int z  = blockIdx.z;
    const int b  = z >> 1;
    const int br = z & 1;
    const _Float16* A    = br ? Wc16 : Wso16;
    const float*    X    = (br ? xc : xso) + (size_t)b * CIN * NQ;
    const float*    bias = br ? bc : bso;
    _Float16*       Out  = fbuf + (size_t)b * NCAT * LDF + (size_t)br * NCH * LDF;
    gemm_core<true, true>(A, X, bias, Out, CIN, NQ, LDF, blockIdx.x, blockIdx.y);
}

// GEMM2: f16 x f16 -> f32 y-buffer (bn1/bn2 folded into W1'' and bias)
__global__ __launch_bounds__(256) void gemm2_kernel(
    const _Float16* __restrict__ W116, const _Float16* __restrict__ fbuf,
    const float* __restrict__ b1f, float* __restrict__ ybuf)
{
    const int b = blockIdx.z;
    const _Float16* Bsrc = fbuf + (size_t)b * NCAT * LDF;
    float*          Out  = ybuf + (size_t)b * NH * NQ;
    gemm_core<false, false>(W116, Bsrc, b1f, Out, NCAT, LDF, NQ, blockIdx.x, blockIdx.y);
}

// Final: out[b,n] = ( (1/49) * sum_{o,hw} W2[o]*y[b,o,n*49+hw] + b2 > 0 ) ? 1 : 0
// (sigmoid(t) > 0.5  <=>  t > 0)
__global__ __launch_bounds__(256) void final_kernel(
    const float* __restrict__ ybuf, const float* __restrict__ W2,
    const float* __restrict__ b2, float* __restrict__ out)
{
    const int bn = blockIdx.x;
    const int b  = bn / NSEG;
    const int n  = bn - b * NSEG;
    const float* yb = ybuf + (size_t)b * NH * NQ + n * HW;
    float partial = 0.f;
    for (int idx = threadIdx.x; idx < NH * HW; idx += 256) {
        int o  = idx / HW;
        int hw = idx - o * HW;
        partial += W2[o] * yb[(size_t)o * NQ + hw];
    }
    float tot = block_reduce_sum(partial);
    if (threadIdx.x == 0) {
        float t = tot * (1.0f / 49.0f) + b2[0];
        out[bn] = (t > 0.f) ? 1.0f : 0.0f;
    }
}

// ---------------------------------------------------------------------------
extern "C" void kernel_launch(void* const* d_in, const int* in_sizes, int n_in,
                              void* d_out, int out_size, void* d_ws, size_t ws_size,
                              hipStream_t stream)
{
    (void)in_sizes; (void)n_in; (void)out_size; (void)ws_size;

    const float* x_so   = (const float*)d_in[0];
    const float* x_c    = (const float*)d_in[1];
    const float* bnso_g = (const float*)d_in[2];
    const float* bnso_b = (const float*)d_in[3];
    const float* bnso_m = (const float*)d_in[4];
    const float* bnso_v = (const float*)d_in[5];
    const float* W_so   = (const float*)d_in[6];
    const float* b_so   = (const float*)d_in[7];
    const float* bnc_g  = (const float*)d_in[8];
    const float* bnc_b  = (const float*)d_in[9];
    const float* bnc_m  = (const float*)d_in[10];
    const float* bnc_v  = (const float*)d_in[11];
    const float* W_c    = (const float*)d_in[12];
    const float* b_c    = (const float*)d_in[13];
    const float* bn1_g  = (const float*)d_in[14];
    const float* bn1_b  = (const float*)d_in[15];
    const float* bn1_m  = (const float*)d_in[16];
    const float* bn1_v  = (const float*)d_in[17];
    const float* W1     = (const float*)d_in[18];
    const float* b1     = (const float*)d_in[19];
    const float* bn2_g  = (const float*)d_in[20];
    const float* bn2_b  = (const float*)d_in[21];
    const float* bn2_m  = (const float*)d_in[22];
    const float* bn2_v  = (const float*)d_in[23];
    const float* W2     = (const float*)d_in[24];
    const float* b2     = (const float*)d_in[25];

    // workspace carve-up
    char* ws = (char*)d_ws;
    size_t off = 0;
    _Float16* Wso16 = (_Float16*)(ws + off); off += (size_t)NCH * CIN * sizeof(_Float16);
    _Float16* Wc16  = (_Float16*)(ws + off); off += (size_t)NCH * CIN * sizeof(_Float16);
    _Float16* W116  = (_Float16*)(ws + off); off += (size_t)NH * NCAT * sizeof(_Float16);
    float*    bsoF  = (float*)(ws + off);    off += (size_t)NCH * sizeof(float);
    float*    bcF   = (float*)(ws + off);    off += (size_t)NCH * sizeof(float);
    float*    b1F   = (float*)(ws + off);    off += (size_t)NH * sizeof(float);
    _Float16* fbuf  = (_Float16*)(ws + off); off += (size_t)BQ * NCAT * LDF * sizeof(_Float16);
    float*    ybuf  = (float*)(ws + off);    off += (size_t)BQ * NH * NQ * sizeof(float);

    // 1) fold BN into f16 weights / f32 biases
    fold_branch_kernel<<<NCH, 256, 0, stream>>>(W_so, bnso_g, bnso_b, bnso_m, bnso_v,
                                                b_so, Wso16, bsoF, CIN);
    fold_branch_kernel<<<NCH, 256, 0, stream>>>(W_c, bnc_g, bnc_b, bnc_m, bnc_v,
                                                b_c, Wc16, bcF, CIN);
    fold1_kernel<<<NH, 256, 0, stream>>>(W1, bn1_g, bn1_b, bn1_m, bn1_v, b1,
                                         bn2_g, bn2_b, bn2_m, bn2_v, W116, b1F);

    // 2) branch GEMMs + lrelu -> f16 concat buffer
    dim3 g1(NCH / 128, (NQ + 63) / 64, BQ * 2);
    gemm1_kernel<<<g1, 256, 0, stream>>>(Wso16, Wc16, x_so, x_c, bsoF, bcF, fbuf);

    // 3) hidden GEMM + lrelu -> f32 y buffer
    dim3 g2(NH / 128, (NQ + 63) / 64, BQ);
    gemm2_kernel<<<g2, 256, 0, stream>>>(W116, fbuf, b1F, ybuf);

    // 4) final dot + spatial mean + sigmoid threshold
    final_kernel<<<BQ * NSEG, 256, 0, stream>>>(ybuf, W2, b2, (float*)d_out);
}
